// CustomLLM_63471026700617
// MI455X (gfx1250) — matrix operand order repaired and checked by reading, not compiled, not run on hardware
//
#include <hip/hip_runtime.h>
#include <hip/hip_bf16.h>
#include <math.h>

// GPT-2-small forward, CDNA5 (gfx1250), wave32 + WMMA bf16.
#define B_  2
#define T_  1024
#define D_  768
#define H_  12
#define HD_ 64
#define L_  12
#define V_  32000
#define M_  (B_ * T_)   // 2048 token rows

typedef __attribute__((ext_vector_type(16))) __bf16 bf16x16;
typedef __attribute__((ext_vector_type(8)))  __bf16 bf16x8;
typedef __attribute__((ext_vector_type(4)))  __bf16 bf16x4;
typedef __attribute__((ext_vector_type(2)))  __bf16 bf16x2;
typedef __attribute__((ext_vector_type(8)))  float  f32x8;
typedef __attribute__((ext_vector_type(4)))  float  f32x4;

union FragB { bf16x16 v; bf16x8 q[2]; __bf16 h[16]; };

// ---------------------------------------------------------------------------
// Embedding: x[row][d] = wte[idx[row]][d] + wpe[row % T][d]
// ---------------------------------------------------------------------------
__global__ void embed_k(const int* __restrict__ idx, const float* __restrict__ wte,
                        const float* __restrict__ wpe, float* __restrict__ x)
{
  int row = blockIdx.x;
  int t   = row & (T_ - 1);
  int tok = idx[row];
  const float* we = wte + (size_t)tok * D_;
  const float* wp = wpe + (size_t)t * D_;
  float* xr = x + (size_t)row * D_;
  for (int d = threadIdx.x; d < D_; d += blockDim.x) xr[d] = we[d] + wp[d];
}

// ---------------------------------------------------------------------------
// LayerNorm (fp32 in -> bf16 out). One wave per row of 768 (24 elems/lane).
// ---------------------------------------------------------------------------
__global__ void ln_bf16(const float* __restrict__ x, const float* __restrict__ w,
                        const float* __restrict__ b, __bf16* __restrict__ out,
                        int M)
{
  int row = blockIdx.x * (blockDim.x >> 5) + (threadIdx.x >> 5);
  if (row >= M) return;
  int lane = threadIdx.x & 31;
  const float* xr = x + (size_t)row * D_;
  float v[24];
  float s = 0.f;
  #pragma unroll
  for (int j = 0; j < 24; ++j) { v[j] = xr[lane + j * 32]; s += v[j]; }
  #pragma unroll
  for (int m = 16; m; m >>= 1) s += __shfl_xor(s, m, 32);
  float mean = s * (1.f / 768.f);
  float q = 0.f;
  #pragma unroll
  for (int j = 0; j < 24; ++j) { float d = v[j] - mean; q += d * d; }
  #pragma unroll
  for (int m = 16; m; m >>= 1) q += __shfl_xor(q, m, 32);
  float rstd = rsqrtf(q * (1.f / 768.f) + 1e-5f);
  __bf16* orow = out + (size_t)row * D_;
  #pragma unroll
  for (int j = 0; j < 24; ++j) {
    int c = lane + j * 32;
    orow[c] = (__bf16)((v[j] - mean) * rstd * w[c] + b[c]);
  }
}

// ---------------------------------------------------------------------------
// GEMM: C[M,N] = A_bf16[M,K] x W_f32->bf16[K,N] (+bias) (+gelu) (+residual)
// Block = 256 threads (8 waves), 128x128 macro-tile; each wave does 16x128
// (8 accumulators, A fragment reused 8x). Weight tile (32Kx128N) is staged
// into LDS *transposed* (col-major, padded stride 72 ushorts = 144 B) so each
// lane's B fragment is one contiguous 32 B read (2x ds_load_b128, bank-clean).
// wT=1: W stored [N,K] row-major (tied lm_head with wte).
// ---------------------------------------------------------------------------
#define LDT 72   // padded row stride (ushorts) of transposed LDS tile
__global__ void gemm_wmma(const __bf16* __restrict__ A,
                          const float* __restrict__ W,
                          const float* __restrict__ bias,
                          const float* __restrict__ resid,
                          float* __restrict__ outF,
                          __bf16* __restrict__ outB,
                          int M, int N, int K, int wT, int gelu)
{
  __shared__ __bf16 tileBT[128][LDT];        // 18 KB: [col][K] bf16
  const int tid  = threadIdx.x;
  const int wave = tid >> 5;
  const int lane = tid & 31;
  const int half = lane >> 4;
  const int l16  = lane & 15;
  const int n0   = blockIdx.x * 128;
  const int m0   = blockIdx.y * 128;
  const int tmB  = m0 + wave * 16;           // this wave's 16 output rows
  const int rowA = tmB + l16;                // A-matrix row handled by this lane

  f32x8 acc[8] = {};
  const __bf16* arow = A + (size_t)rowA * K;

  for (int k0 = 0; k0 < K; k0 += 32) {
    // ---- cooperative stage of W[k0:k0+32, n0:n0+128] into LDS (transposed) ----
    if (wT) {
      // W is [N,K]: 2 threads per column, 16 contiguous k each. Contiguous stores.
      int n  = tid >> 1;
      int kb = (tid & 1) * 16;
      const float* src = W + (size_t)(n0 + n) * K + k0 + kb;
      float tmp[16];
      #pragma unroll
      for (int j = 0; j < 4; ++j) *(f32x4*)(tmp + 4 * j) = *(const f32x4*)(src + 4 * j);
      if (k0 + 32 < K) __builtin_prefetch(src + 32, 0, 1);
      #pragma unroll
      for (int g = 0; g < 2; ++g) {
        bf16x8 pk;
        #pragma unroll
        for (int e = 0; e < 8; ++e) pk[e] = (__bf16)tmp[g * 8 + e];
        *(bf16x8*)&tileBT[n][kb + g * 8] = pk;
      }
    } else {
      // W is [K,N]: thread -> 4 k rows x 4 n cols; transpose in registers,
      // write bf16x4 (4 consecutive k) per column.
      int kq = (tid & 7) * 4;
      int nb = (tid >> 3) * 4;
      float vals[4][4];
      #pragma unroll
      for (int j = 0; j < 4; ++j) {
        const float* src = W + (size_t)(k0 + kq + j) * N + n0 + nb;
        *(f32x4*)(vals[j]) = *(const f32x4*)(src);
        if (k0 + 32 < K) __builtin_prefetch(src + (size_t)32 * N, 0, 1);
      }
      #pragma unroll
      for (int n = 0; n < 4; ++n) {
        bf16x4 pk;
        #pragma unroll
        for (int j = 0; j < 4; ++j) pk[j] = (__bf16)vals[j][n];
        *(bf16x4*)&tileBT[nb + n][kq] = pk;
      }
    }
    __syncthreads();

    // ---- A fragment: two contiguous 8-element runs (2x 16 B global loads) ----
    FragB a;
    a.q[0] = *(const bf16x8*)(arow + k0 + (half << 4));
    a.q[1] = *(const bf16x8*)(arow + k0 + 8 + (half << 4));

    // ---- 8 WMMAs; B fragment = one contiguous 32 B LDS read per lane ----
    #pragma unroll
    for (int nt = 0; nt < 8; ++nt) {
      FragB b;
      b.v = *(const bf16x16*)&tileBT[nt * 16 + l16][half << 4];
      acc[nt] = __builtin_amdgcn_wmma_f32_16x16x32_bf16(false, a.v, false, b.v,
                                                        (short)0, acc[nt], false, false);
    }
    __syncthreads();
  }

  // ---- epilogue: C/D layout row = half*8 + r, col = l16 ----
  #pragma unroll
  for (int nt = 0; nt < 8; ++nt) {
    int ng = n0 + nt * 16 + l16;
    #pragma unroll
    for (int r = 0; r < 8; ++r) {
      int mg = tmB + (half << 3) + r;
      float v = acc[nt][r];
      if (bias) v += bias[ng];
      if (gelu) v = 0.5f * v * (1.0f + erff(v * 0.70710678118f));
      size_t o = (size_t)mg * N + ng;
      if (resid) v += resid[o];
      if (outB) outB[o] = (__bf16)v; else outF[o] = v;
    }
  }
}

// ---------------------------------------------------------------------------
// Causal attention, flash-style, WMMA bf16, qkv stored bf16.
// One block (256 thr / 8 waves) = one (b,h) x 128 q-rows; wave w owns q-tile
// qb = tb*8+w. Per 32-key chunk:
//   - K tile staged global->LDS with GLOBAL_LOAD_ASYNC_TO_LDS_B128 (pure byte
//     move, tracked by ASYNCcnt) -- no VGPR round-trip.
//   - V tile staged transposed (vdim-major) via register packing.
// Every WMMA B fragment is a single contiguous 32 B LDS read. The 1/sqrt(64)
// scale is applied to f32 scores (exact). Causal skipping is wave-uniform.
// ---------------------------------------------------------------------------
#define LDK 72   // padded stride of K tile rows (ushorts)
#define LDV 40   // padded stride of V^T tile rows (ushorts)
__global__ void attn_wmma(const __bf16* __restrict__ qkv, __bf16* __restrict__ y)
{
  __shared__ __bf16 klds[32][LDK];           // 4.5 KB: [key][head-dim]
  __shared__ __bf16 vlds[64][LDV];           // 5 KB:   [head-dim][key] (transposed)
  __shared__ __bf16 plds[8][512];            // 8 KB:   per-wave 16x32 P staging
  const int tid  = threadIdx.x;
  const int wave = tid >> 5;
  const int lane = tid & 31;
  const int half = lane >> 4;
  const int l16  = lane & 15;
  int bid = blockIdx.x;                      // 0 .. B*H*(T/128)-1
  int b   = bid / (H_ * 8);
  int rem = bid % (H_ * 8);
  int h   = rem / 8;
  int tb  = rem % 8;
  int qb  = tb * 8 + wave;                   // this wave's 16-row q tile
  __bf16* my = plds[wave];

  // Q fragments (bf16 source): two contiguous 8-element runs per 32-dim chunk.
  FragB aq0, aq1;
  {
    const __bf16* qr = qkv + (size_t)(b * T_ + qb * 16 + l16) * (3 * D_) + h * HD_;
    aq0.q[0] = *(const bf16x8*)(qr + (half << 4));
    aq0.q[1] = *(const bf16x8*)(qr + 8 + (half << 4));
    aq1.q[0] = *(const bf16x8*)(qr + 32 + (half << 4));
    aq1.q[1] = *(const bf16x8*)(qr + 40 + (half << 4));
  }

  // Per-chunk staging addresses (constant strides; only key row changes).
  const int kkey = tid >> 3, kseg = tid & 7; // K: 32 keys x 8 segs of 8 dims
  const unsigned klds_off =
      (unsigned)(unsigned long)&klds[kkey][kseg * 8];
  const int vk2 = (tid & 15) * 2, vdq = (tid >> 4) * 4;  // V: 16 key-pairs x 16 dim-quads

  float mrow[8], lrow[8];
  f32x8 acc[4] = {};
  #pragma unroll
  for (int r = 0; r < 8; ++r) { mrow[r] = -3.4e38f; lrow[r] = 0.f; }

  int nchunk = tb * 4 + 4;                   // chunks needed by the block's last q row
  for (int c = 0; c < nchunk; ++c) {
    int j0 = c * 32;
    // ---- K tile: async byte-move global -> LDS (16 B per lane) ----
    {
      const __bf16* kp = qkv + (size_t)(b * T_ + j0 + kkey) * (3 * D_) + D_ + h * HD_
                         + kseg * 8;
      asm volatile("global_load_async_to_lds_b128 %0, %1, off"
                   :: "v"(klds_off), "v"(kp) : "memory");
    }
    // ---- V tile: stage transposed (vdim-major) with bf16x2 packing ----
    {
      const __bf16* vp0 = qkv + (size_t)(b * T_ + j0 + vk2) * (3 * D_) + 2 * D_ + h * HD_
                          + vdq;
      const __bf16* vp1 = vp0 + 3 * D_;
      bf16x4 a0 = *(const bf16x4*)vp0;
      bf16x4 a1 = *(const bf16x4*)vp1;
      #pragma unroll
      for (int e = 0; e < 4; ++e) {
        bf16x2 pr; pr[0] = a0[e]; pr[1] = a1[e];
        *(bf16x2*)&vlds[vdq + e][vk2] = pr;
      }
    }
    asm volatile("s_wait_asynccnt 0x0" ::: "memory");
    __syncthreads();

    if (j0 < qb * 16 + 16) {                 // wave-uniform causal skip
      f32x8 s0 = {}, s1 = {};
      FragB bk;
      bk.v = *(const bf16x16*)&klds[l16][half << 4];            // keys j0..j0+15
      s0 = __builtin_amdgcn_wmma_f32_16x16x32_bf16(false, aq0.v, false, bk.v, (short)0, s0, false, false);
      bk.v = *(const bf16x16*)&klds[l16][32 + (half << 4)];
      s0 = __builtin_amdgcn_wmma_f32_16x16x32_bf16(false, aq1.v, false, bk.v, (short)0, s0, false, false);
      bk.v = *(const bf16x16*)&klds[16 + l16][half << 4];       // keys j0+16..j0+31
      s1 = __builtin_amdgcn_wmma_f32_16x16x32_bf16(false, aq0.v, false, bk.v, (short)0, s1, false, false);
      bk.v = *(const bf16x16*)&klds[16 + l16][32 + (half << 4)];
      s1 = __builtin_amdgcn_wmma_f32_16x16x32_bf16(false, aq1.v, false, bk.v, (short)0, s1, false, false);

      // Online softmax per score row (row = half*8+r lives on 16 lanes of this half).
      #pragma unroll
      for (int r = 0; r < 8; ++r) {
        int qr_ = qb * 16 + (half << 3) + r;
        float v0 = (j0 + l16      <= qr_) ? s0[r] * 0.125f : -3.4e38f;
        float v1 = (j0 + 16 + l16 <= qr_) ? s1[r] * 0.125f : -3.4e38f;
        float mx = fmaxf(v0, v1);
        #pragma unroll
        for (int msk = 8; msk; msk >>= 1) mx = fmaxf(mx, __shfl_xor(mx, msk, 32));
        float mn = fmaxf(mrow[r], mx);
        float sc = expf(mrow[r] - mn);
        float p0 = expf(v0 - mn);
        float p1 = expf(v1 - mn);
        float rs = p0 + p1;
        #pragma unroll
        for (int msk = 8; msk; msk >>= 1) rs += __shfl_xor(rs, msk, 32);
        lrow[r] = lrow[r] * sc + rs;
        mrow[r] = mn;
        #pragma unroll
        for (int n = 0; n < 4; ++n) acc[n][r] *= sc;
        int prow = (half << 3) + r;          // stage P (C-layout) into LDS
        my[prow * 32 + l16]      = (__bf16)p0;
        my[prow * 32 + 16 + l16] = (__bf16)p1;
      }
      asm volatile("s_wait_dscnt 0x0" ::: "memory");   // same-wave LDS RAW
      FragB pa;                               // re-read P in A-layout: 2x 16 B reads
      pa.q[0] = *(const bf16x8*)&my[l16 * 32 + (half << 4)];
      pa.q[1] = *(const bf16x8*)&my[l16 * 32 + 8 + (half << 4)];
      asm volatile("" ::: "memory");
      #pragma unroll
      for (int n = 0; n < 4; ++n) {           // O += P(16x32) x V(32x64)
        FragB bv;
        bv.v = *(const bf16x16*)&vlds[n * 16 + l16][half << 4];
        acc[n] = __builtin_amdgcn_wmma_f32_16x16x32_bf16(false, pa.v, false, bv.v,
                                                         (short)0, acc[n], false, false);
      }
    }
    __syncthreads();
  }

  #pragma unroll
  for (int n = 0; n < 4; ++n)
    #pragma unroll
    for (int r = 0; r < 8; ++r) {
      int qr_ = qb * 16 + (half << 3) + r;
      float o = acc[n][r] / lrow[r];
      y[(size_t)(b * T_ + qr_) * D_ + h * HD_ + n * 16 + l16] = (__bf16)o;
    }
}

// ---------------------------------------------------------------------------
// Loss: per-row logsumexp NLL, then deterministic single-block reduction.
// ---------------------------------------------------------------------------
__global__ void nll_rows(const float* __restrict__ logits, const int* __restrict__ tgt,
                         float* __restrict__ nll)
{
  __shared__ float red[256];
  int row = blockIdx.x;
  const float* lr = logits + (size_t)row * V_;
  int tid = threadIdx.x;
  float mx = -3.4e38f;
  for (int i = tid; i < V_; i += 256) mx = fmaxf(mx, lr[i]);
  red[tid] = mx; __syncthreads();
  for (int s = 128; s; s >>= 1) { if (tid < s) red[tid] = fmaxf(red[tid], red[tid + s]); __syncthreads(); }
  float m = red[0]; __syncthreads();
  float sum = 0.f;
  for (int i = tid; i < V_; i += 256) sum += expf(lr[i] - m);
  red[tid] = sum; __syncthreads();
  for (int s = 128; s; s >>= 1) { if (tid < s) red[tid] += red[tid + s]; __syncthreads(); }
  if (tid == 0) {
    float lse = m + logf(red[0]);
    int t = tgt[row];
    nll[row] = (t != -1) ? (lse - lr[t < 0 ? 0 : t]) : 0.f;
  }
}

__global__ void loss_final(const float* __restrict__ nll, const int* __restrict__ tgt,
                           float* __restrict__ out)
{
  __shared__ float red[256];
  __shared__ float cnt[256];
  int tid = threadIdx.x;
  float s = 0.f, c = 0.f;
  for (int i = tid; i < M_; i += 256) { s += nll[i]; c += (tgt[i] != -1) ? 1.f : 0.f; }
  red[tid] = s; cnt[tid] = c; __syncthreads();
  for (int k = 128; k; k >>= 1) {
    if (tid < k) { red[tid] += red[tid + k]; cnt[tid] += cnt[tid + k]; }
    __syncthreads();
  }
  if (tid == 0) out[0] = red[0] / fmaxf(cnt[0], 1.f);
}

// ---------------------------------------------------------------------------
extern "C" void kernel_launch(void* const* d_in, const int* in_sizes, int n_in,
                              void* d_out, int out_size, void* d_ws, size_t ws_size,
                              hipStream_t stream)
{
  (void)in_sizes; (void)n_in; (void)out_size; (void)ws_size;
  const int*   idx   = (const int*)d_in[0];
  const int*   tgt   = (const int*)d_in[1];
  const float* wte   = (const float*)d_in[2];
  const float* wpe   = (const float*)d_in[3];
  const float* ln1w  = (const float*)d_in[4];
  const float* ln1b  = (const float*)d_in[5];
  const float* attnw = (const float*)d_in[6];
  const float* attnb = (const float*)d_in[7];
  const float* projw = (const float*)d_in[8];
  const float* projb = (const float*)d_in[9];
  const float* ln2w  = (const float*)d_in[10];
  const float* ln2b  = (const float*)d_in[11];
  const float* fc1w  = (const float*)d_in[12];
  const float* fc1b  = (const float*)d_in[13];
  const float* fc2w  = (const float*)d_in[14];
  const float* fc2b  = (const float*)d_in[15];
  const float* lnfw  = (const float*)d_in[16];
  const float* lnfb  = (const float*)d_in[17];

  char* ws = (char*)ws ? (char*)d_ws : (char*)d_ws;
  ws = (char*)d_ws;
  float*  x     = (float*)ws;   ws += (size_t)M_ * D_ * 4;       // residual stream f32
  __bf16* lnbf  = (__bf16*)ws;  ws += (size_t)M_ * D_ * 2;       // LN output bf16
  __bf16* qkvbf = (__bf16*)ws;  ws += (size_t)M_ * 3 * D_ * 2;   // qkv bf16
  __bf16* ybf   = (__bf16*)ws;  ws += (size_t)M_ * D_ * 2;       // attn out bf16
  __bf16* hbf   = (__bf16*)ws;  ws += (size_t)M_ * 4 * D_ * 2;   // gelu(fc1) bf16
  float*  nll   = (float*)ws;   ws += (size_t)M_ * 4;

  float* logits = (float*)d_out;
  float* loss   = logits + (size_t)M_ * V_;

  embed_k<<<M_, 256, 0, stream>>>(idx, wte, wpe, x);

  auto gemm = [&](const __bf16* A, const float* W, const float* bias,
                  const float* resid, float* oF, __bf16* oB,
                  int M, int N, int K, int wT, int gelu) {
    dim3 grid(N / 128, M / 128);             // 128x128 macro-tile per 8-wave block
    gemm_wmma<<<grid, 256, 0, stream>>>(A, W, bias, resid, oF, oB, M, N, K, wT, gelu);
  };

  for (int l = 0; l < L_; ++l) {
    ln_bf16<<<M_ / 8, 256, 0, stream>>>(x, ln1w + l * D_, ln1b + l * D_, lnbf, M_);
    gemm(lnbf, attnw + (size_t)l * D_ * 3 * D_, attnb + (size_t)l * 3 * D_,
         nullptr, nullptr, qkvbf, M_, 3 * D_, D_, 0, 0);        // qkv -> bf16
    attn_wmma<<<B_ * H_ * (T_ / 128), 256, 0, stream>>>(qkvbf, ybf);
    gemm(ybf, projw + (size_t)l * D_ * D_, projb + (size_t)l * D_,
         x, x, nullptr, M_, D_, D_, 0, 0);                      // x += proj(y)
    ln_bf16<<<M_ / 8, 256, 0, stream>>>(x, ln2w + l * D_, ln2b + l * D_, lnbf, M_);
    gemm(lnbf, fc1w + (size_t)l * D_ * 4 * D_, fc1b + (size_t)l * 4 * D_,
         nullptr, nullptr, hbf, M_, 4 * D_, D_, 0, 1);          // gelu -> bf16
    gemm(hbf, fc2w + (size_t)l * 4 * D_ * D_, fc2b + (size_t)l * D_,
         x, x, nullptr, M_, D_, 4 * D_, 0, 0);                  // x += fc2(h)
  }
  ln_bf16<<<M_ / 8, 256, 0, stream>>>(x, lnfw, lnfb, lnbf, M_);
  gemm(lnbf, wte, nullptr, nullptr, logits, nullptr, M_, V_, D_, 1, 0);  // tied lm_head (W^T)
  nll_rows<<<M_, 256, 0, stream>>>(logits, tgt, nll);
  loss_final<<<1, 256, 0, stream>>>(nll, tgt, loss);
}
